// InputsConcatConditionalDecoder_66151086293261
// MI455X (gfx1250) — compile-verified, matrix operand order checked by reading, not run on hardware
//
#include <hip/hip_runtime.h>
#include <hip/hip_bf16.h>
#include <cstdint>

// ---------------------------------------------------------------------------
// InputsConcatConditionalDecoder for MI455X (gfx1250, wave32, WMMA)
//
//  * bf16 weights (~34MB, L2-resident), f32 accumulation via
//    v_wmma_f32_16x16x32_bf16.
//  * Single fused K-loop (x|h concat, K=2048), register double-buffering of
//    weight fragments; raw s_barrier_signal/-wait (no loadcnt flush) so the
//    weight loads pipeline across the barrier.
//  * A-tile staged through triple-buffered LDS with
//    global_load_async_to_lds_b64 (prefetch depth 2, s_wait_asynccnt 0x2),
//    read back with ds_load_b128.  One barrier per K-iteration.
//  * LSTM gate math fused into the GEMM epilogue.
// ---------------------------------------------------------------------------

#define B_   512
#define H_   1024
#define L_   2
#define T_   144
#define COND_ 32
#define CONDH_ 512
#define EMB_ 511          // H - CONDH - 1
#define OUT_ 17
#define BH_  (B_ * H_)    // 524288
#define W4HH_ ((size_t)4 * H_ * H_)   // 4,194,304 elements per matrix

typedef __attribute__((ext_vector_type(8)))  unsigned short v8us;
typedef __attribute__((ext_vector_type(16))) unsigned short v16us;
typedef __attribute__((ext_vector_type(16))) __bf16        v16bf;
typedef __attribute__((ext_vector_type(8)))  float          v8f;

__device__ __forceinline__ unsigned short f2bf(float f) {
  unsigned int u = __float_as_uint(f);
  u = (u + 0x7FFFu + ((u >> 16) & 1u)) >> 16;   // round-to-nearest-even
  return (unsigned short)u;
}
__device__ __forceinline__ float bf2f(unsigned short u) {
  return __uint_as_float(((unsigned int)u) << 16);
}
__device__ __forceinline__ float sigm(float x) { return 1.0f / (1.0f + __expf(-x)); }

__device__ __forceinline__ v16bf pack_frag(v8us lo, v8us hi) {
  v16us w = __builtin_shufflevector(lo, hi, 0,1,2,3,4,5,6,7,8,9,10,11,12,13,14,15);
  union { v16us u; v16bf b; } cvt; cvt.u = w;
  return cvt.b;
}

// 16x32 bf16 WMMA fragment from a row-major matrix (A layout; B identical with
// lane=col).  lane m: V0-3 = K{0..7}(lo)/{8..15}(hi), V4-7 = +16.
__device__ __forceinline__ v16bf load_frag(const unsigned short* __restrict__ base,
                                           int ld, int row0, int k0) {
  const int lane = threadIdx.x & 31;
  const unsigned short* p =
      base + (size_t)(row0 + (lane & 15)) * (size_t)ld + (size_t)(k0 + ((lane >> 4) << 3));
  v8us lo = *(const v8us*)p;
  v8us hi = *(const v8us*)(p + 16);
  return pack_frag(lo, hi);
}

// workgroup split-barrier WITHOUT the loadcnt/dscnt flush of __syncthreads()
__device__ __forceinline__ void wg_barrier() {
  asm volatile("s_barrier_signal -1\n\ts_barrier_wait -1" ::: "memory");
}

// ---------------------------------------------------------------------------
// One LSTM layer, one timestep, fully fused.
// Grid: (H/64, B/32), block 256 = 8 waves.
// ---------------------------------------------------------------------------
__global__ __launch_bounds__(256)
void lstm_cell(const unsigned short* __restrict__ X,      // (B,H) bf16
               const unsigned short* __restrict__ Hprev,  // (B,H) bf16
               const unsigned short* __restrict__ Wih,    // (4H,H) bf16
               const unsigned short* __restrict__ Whh,    // (4H,H) bf16
               const float* __restrict__ bias,            // (4H) bih+bhh
               float* __restrict__ C,                     // (B,H) f32 in/out
               unsigned short* __restrict__ Hnext) {      // (B,H) bf16 out
  __shared__ unsigned short sA[3][32 * 32];   // triple-buffered 32x32 bf16 A tile

  const int tid  = threadIdx.x;
  const int wave = tid >> 5;
  const int lane = tid & 31;
  const int mt   = wave >> 2;                 // 0..1  (batch sub-tile)
  const int r0b  = blockIdx.y * 32;           // block batch-row base
  const int r0   = r0b + mt * 16;
  const int j0   = blockIdx.x * 64 + (wave & 3) * 16;

  // async A-tile stage: each of 256 threads copies 8 bytes (4 halves)
  const int arow = tid >> 3;                  // 0..31
  const int acol = (tid & 7) * 4;             // half index 0..28

  const unsigned ldsBase    = (unsigned)(size_t)(&sA[0][0]);
  const unsigned myStoreOff = (unsigned)((arow * 32 + acol) * 2);
  const unsigned myReadOff  =
      (unsigned)(((mt * 16 + (lane & 15)) * 32 + ((lane >> 4) << 3)) * 2);

  auto issueA = [&](int kk, unsigned bufOff) {
    const unsigned short* src = (kk < H_) ? X : Hprev;
    const int k0 = kk & (H_ - 1);
    const unsigned short* g = src + (size_t)(r0b + arow) * H_ + (size_t)(k0 + acol);
    unsigned ldsoff = ldsBase + bufOff + myStoreOff;
    asm volatile("global_load_async_to_lds_b64 %0, %1, off"
                 :: "v"(ldsoff), "v"(g) : "memory");
  };
  auto readA = [&](unsigned bufOff) -> v16bf {
    const unsigned short* p =
        (const unsigned short*)((const char*)(&sA[0][0]) + bufOff + myReadOff);
    v8us lo = *(const v8us*)p;
    v8us hi = *(const v8us*)(p + 16);
    return pack_frag(lo, hi);
  };
  auto wsel = [&](int kk) { return (kk < H_) ? Wih : Whh; };

  v8f acc0 = {}, acc1 = {}, acc2 = {}, acc3 = {};

  // rotating LDS buffer byte-offsets (avoid %3 in the loop)
  unsigned off0 = 0, off1 = 2048, off2 = 4096;

  // ---- pipeline prologue: stage tiles 0 and 1, weight frags for kk=0
  issueA(0,  off0);
  issueA(32, off1);
  v16bf b0 = load_frag(wsel(0), H_, 0 * H_ + j0, 0);
  v16bf b1 = load_frag(wsel(0), H_, 1 * H_ + j0, 0);
  v16bf b2 = load_frag(wsel(0), H_, 2 * H_ + j0, 0);
  v16bf b3 = load_frag(wsel(0), H_, 3 * H_ + j0, 0);

  for (int kk = 0; kk < 2 * H_; kk += 32) {
    // stage tile kk+64 (clamped dummy near the tail keeps the wait constant)
    const int kn2 = (kk + 64 < 2 * H_) ? kk + 64 : kk;
    issueA(kn2, off2);

    // next iteration's weight fragments (register double buffer)
    const int kn = (kk + 32 < 2 * H_) ? kk + 32 : kk;
    const unsigned short* Wn = wsel(kn);
    const int k0n = kn & (H_ - 1);
    v16bf nb0 = load_frag(Wn, H_, 0 * H_ + j0, k0n);
    v16bf nb1 = load_frag(Wn, H_, 1 * H_ + j0, k0n);
    v16bf nb2 = load_frag(Wn, H_, 2 * H_ + j0, k0n);
    v16bf nb3 = load_frag(Wn, H_, 3 * H_ + j0, k0n);

    // tile kk resident: only the two newer tile-ops may still be in flight
    asm volatile("s_wait_asynccnt 0x2" ::: "memory");
    wg_barrier();

    v16bf a = readA(off0);
    acc0 = __builtin_amdgcn_wmma_f32_16x16x32_bf16(false, a, false, b0, (short)0, acc0, false, false);
    acc1 = __builtin_amdgcn_wmma_f32_16x16x32_bf16(false, a, false, b1, (short)0, acc1, false, false);
    acc2 = __builtin_amdgcn_wmma_f32_16x16x32_bf16(false, a, false, b2, (short)0, acc2, false, false);
    acc3 = __builtin_amdgcn_wmma_f32_16x16x32_bf16(false, a, false, b3, (short)0, acc3, false, false);

    b0 = nb0; b1 = nb1; b2 = nb2; b3 = nb3;
    unsigned tmp = off0; off0 = off1; off1 = off2; off2 = tmp;   // rotate buffers
  }

  // ---- fused LSTM epilogue.  C/D layout: VGPR v -> row v (+8 hi lanes),
  // lane&15 -> column.
  const int col = j0 + (lane & 15);
  const int rhi = (lane >> 4) * 8;
  const float bi = bias[0 * H_ + col];
  const float bf = bias[1 * H_ + col];
  const float bg = bias[2 * H_ + col];
  const float bo = bias[3 * H_ + col];
#pragma unroll
  for (int v = 0; v < 8; ++v) {
    const int row = r0 + rhi + v;
    const size_t idx = (size_t)row * H_ + col;
    float ig = acc0[v] + bi;
    float fg = acc1[v] + bf;
    float gg = acc2[v] + bg;
    float og = acc3[v] + bo;
    float cn = sigm(fg) * C[idx] + sigm(ig) * tanhf(gg);
    C[idx] = cn;
    Hnext[idx] = f2bf(sigm(og) * tanhf(cn));
  }
}

// ---------------------------------------------------------------------------
// Setup kernels (run once per launch)
// ---------------------------------------------------------------------------
__global__ __launch_bounds__(256)
void conv_weights(const float* __restrict__ Wih, const float* __restrict__ Whh,
                  unsigned short* __restrict__ wih_bf, unsigned short* __restrict__ whh_bf) {
  const size_t n = (size_t)L_ * W4HH_;
  size_t idx = (size_t)blockIdx.x * 256 + threadIdx.x;
  if (idx < n)            wih_bf[idx]     = f2bf(Wih[idx]);
  else if (idx < 2 * n)   whh_bf[idx - n] = f2bf(Whh[idx - n]);
}

__global__ __launch_bounds__(256)
void cond_ff(const float* __restrict__ cond, const float* __restrict__ ffW,
             const float* __restrict__ ffb, unsigned short* __restrict__ condh) {
  int idx = blockIdx.x * 256 + threadIdx.x;
  if (idx >= B_ * CONDH_) return;
  int b = idx >> 9, j = idx & (CONDH_ - 1);
  float s = ffb[j];
#pragma unroll
  for (int k = 0; k < COND_; ++k) s += cond[b * COND_ + k] * ffW[j * COND_ + k];
  condh[idx] = f2bf(s);
}

__global__ __launch_bounds__(256)
void init_state(const float* __restrict__ h0, const float* __restrict__ c0,
                const float* __restrict__ bih, const float* __restrict__ bhh,
                unsigned short* __restrict__ hbuf, float* __restrict__ c_ws,
                float* __restrict__ bias) {
  size_t idx = (size_t)blockIdx.x * 256 + threadIdx.x;
  const size_t nhc = (size_t)L_ * BH_;
  if (idx < nhc) {
    int l = (int)(idx / BH_);
    size_t r = idx - (size_t)l * BH_;
    hbuf[((size_t)l * 2 + 0) * BH_ + r] = f2bf(h0[idx]);   // parity 0 = state at t=0
    c_ws[idx] = c0[idx];
  }
  if (idx < (size_t)L_ * 4 * H_) bias[idx] = bih[idx] + bhh[idx];
}

// x0_t = [emb_table[act_in], dur_in, cond_h]  (bf16, per timestep)
__global__ __launch_bounds__(256)
void build_x0(const int* __restrict__ acts, const float* __restrict__ durs,
              const float* __restrict__ emb, const unsigned short* __restrict__ condh,
              unsigned short* __restrict__ x0, int t) {
  int idx = blockIdx.x * 256 + threadIdx.x;
  if (idx >= BH_) return;
  int b = idx >> 10, col = idx & (H_ - 1);
  unsigned short v;
  if (col < EMB_) {
    int act = (t == 0) ? 0 : acts[b * T_ + t - 1];
    v = f2bf(emb[act * EMB_ + col]);
  } else if (col == EMB_) {
    float d = (t == 0) ? 0.0f : durs[b * T_ + t - 1];
    v = f2bf(d);
  } else {
    v = condh[b * CONDH_ + (col - CONDH_)];
  }
  x0[idx] = v;
}

// pred = h1 @ fc_W^T + fc_b ; out[b,t,:16] = log_softmax ; out[b,t,16] = log_sigmoid
__global__ __launch_bounds__(64)
void fc_head(const unsigned short* __restrict__ H1, const float* __restrict__ fcW,
             const float* __restrict__ fcB, float* __restrict__ out, int t) {
  __shared__ float red[OUT_ * 64];
  const int b = blockIdx.x, tid = threadIdx.x;
  float p[OUT_];
#pragma unroll
  for (int o = 0; o < OUT_; ++o) p[o] = 0.0f;
  for (int k = tid; k < H_; k += 64) {
    float x = bf2f(H1[(size_t)b * H_ + k]);
#pragma unroll
    for (int o = 0; o < OUT_; ++o) p[o] += x * fcW[o * H_ + k];
  }
#pragma unroll
  for (int o = 0; o < OUT_; ++o) red[o * 64 + tid] = p[o];
  __syncthreads();
  if (tid == 0) {
    float pred[OUT_];
    for (int o = 0; o < OUT_; ++o) {
      float s = fcB[o];
      for (int j = 0; j < 64; ++j) s += red[o * 64 + j];
      pred[o] = s;
    }
    float m = pred[0];
    for (int o = 1; o < 16; ++o) m = fmaxf(m, pred[o]);
    float se = 0.0f;
    for (int o = 0; o < 16; ++o) se += __expf(pred[o] - m);
    float lse = m + __logf(se);
    float* dst = out + ((size_t)b * T_ + t) * OUT_;
    for (int o = 0; o < 16; ++o) dst[o] = pred[o] - lse;
    float x = pred[16];
    dst[16] = (x >= 0.0f) ? -log1pf(__expf(-x)) : (x - log1pf(__expf(x)));
  }
}

// ---------------------------------------------------------------------------
extern "C" void kernel_launch(void* const* d_in, const int* in_sizes, int n_in,
                              void* d_out, int out_size, void* d_ws, size_t ws_size,
                              hipStream_t stream) {
  (void)in_sizes; (void)n_in; (void)out_size; (void)ws_size;
  const float* h0   = (const float*)d_in[1];
  const float* c0   = (const float*)d_in[2];
  const float* cond = (const float*)d_in[3];
  const int*   acts = (const int*)d_in[4];
  const float* durs = (const float*)d_in[5];
  const float* emb  = (const float*)d_in[6];
  const float* ffW  = (const float*)d_in[7];
  const float* ffb  = (const float*)d_in[8];
  const float* Wih  = (const float*)d_in[9];
  const float* Whh  = (const float*)d_in[10];
  const float* bih  = (const float*)d_in[11];
  const float* bhh  = (const float*)d_in[12];
  const float* fcW  = (const float*)d_in[13];
  const float* fcb  = (const float*)d_in[14];
  float* out = (float*)d_out;

  // workspace carve-up (~43.5 MB total)
  char* ws = (char*)d_ws;
  size_t off = 0;
  auto alloc = [&](size_t bytes) { size_t o = off; off += (bytes + 255) & ~(size_t)255; return o; };
  unsigned short* wih_bf = (unsigned short*)(ws + alloc(sizeof(unsigned short) * L_ * W4HH_));
  unsigned short* whh_bf = (unsigned short*)(ws + alloc(sizeof(unsigned short) * L_ * W4HH_));
  float*          bias   = (float*)        (ws + alloc(sizeof(float) * L_ * 4 * H_));
  unsigned short* condh  = (unsigned short*)(ws + alloc(sizeof(unsigned short) * B_ * CONDH_));
  unsigned short* x0     = (unsigned short*)(ws + alloc(sizeof(unsigned short) * BH_));
  unsigned short* hbuf   = (unsigned short*)(ws + alloc(sizeof(unsigned short) * 4 * BH_)); // [l][parity]
  float*          c_ws   = (float*)        (ws + alloc(sizeof(float) * L_ * BH_));

  // once-per-launch setup
  {
    size_t n2 = 2 * (size_t)L_ * W4HH_;
    conv_weights<<<(unsigned)((n2 + 255) / 256), 256, 0, stream>>>(Wih, Whh, wih_bf, whh_bf);
    cond_ff<<<(B_ * CONDH_) / 256, 256, 0, stream>>>(cond, ffW, ffb, condh);
    size_t ni = (size_t)L_ * BH_;
    init_state<<<(unsigned)((ni + 255) / 256), 256, 0, stream>>>(h0, c0, bih, bhh, hbuf, c_ws, bias);
  }

  const dim3 gGemm(H_ / 64, B_ / 32);   // 16 x 16 blocks, 8 waves each
  for (int t = 0; t < T_; ++t) {
    const int pr = t & 1, pn = (t + 1) & 1;
    build_x0<<<BH_ / 256, 256, 0, stream>>>(acts, durs, emb, condh, x0, t);
    // layer 0
    lstm_cell<<<gGemm, 256, 0, stream>>>(
        x0, hbuf + (size_t)(0 * 2 + pr) * BH_,
        wih_bf, whh_bf, bias, c_ws,
        hbuf + (size_t)(0 * 2 + pn) * BH_);
    // layer 1 (input = layer 0's fresh hidden state)
    lstm_cell<<<gGemm, 256, 0, stream>>>(
        hbuf + (size_t)(0 * 2 + pn) * BH_, hbuf + (size_t)(2 + pr) * BH_,
        wih_bf + W4HH_, whh_bf + W4HH_, bias + 4 * H_, c_ws + BH_,
        hbuf + (size_t)(2 + pn) * BH_);
    // output head
    fc_head<<<B_, 64, 0, stream>>>(hbuf + (size_t)(2 + pn) * BH_, fcW, fcb, out, t);
  }
}
